// GraphInvariantPointAttention_70987219468954
// MI455X (gfx1250) — compile-verified, matrix operand order checked by reading, not run on hardware
//
#include <hip/hip_runtime.h>
#include <math.h>

// ---------------- problem constants ----------------
#define N_RESC  10000
#define N_EDGEC 320000
#define CS      384
#define CZ      128
#define CH      16
#define NH      12
#define PQn     4
#define PVn     8
#define INF_F   100000.0f
#define NPROJ   1152   // 192(q) + 384(kv) + 144(qp) + 432(kvp)
#define ZCOLS   48     // 12(b) + 32(pair_z) + 4 zero pad
#define FEATD   960    // 192 + 3*96 + 96 + 384

typedef __bf16 bf16_t;
typedef __attribute__((ext_vector_type(16))) __bf16 v16bf;
typedef __attribute__((ext_vector_type(8)))  __bf16 v8bf;
typedef __attribute__((ext_vector_type(8)))  float  v8f;
typedef __attribute__((ext_vector_type(4)))  float  v4f;

// ---------------- WMMA fragment helpers (CDNA5 16x16x32 bf16 layouts) ------
// A (16x32, row-major src): lane m = lane&15, holds k = kb..kb+7 and 16+kb..16+kb+7
// where kb = (lane>>4)*8.  B uses the same indexing on a transposed weight copy
// WT[n][k] (lane supplies n).  16-byte-aligned b128 loads.
__device__ inline v16bf frag_bf16(const bf16_t* base, int ld, int lane) {
    int m  = lane & 15;
    int kb = (lane >> 4) << 3;
    const bf16_t* r = base + m * ld + kb;
    v8bf lo = *(const v8bf*)(r);
    v8bf hi = *(const v8bf*)(r + 16);
    v16bf out;
#pragma unroll
    for (int i = 0; i < 8; ++i) { out[i] = lo[i]; out[8 + i] = hi[i]; }
    return out;
}

// A fragment from fp32 source, converted in-registers to bf16.
// Non-temporal: the z matrix is 164 MB streamed exactly once; keep it out of
// the 192 MB L2 so the edge-pass working set (nodeproj/zp/pts/accumulators,
// ~165 MB) stays resident.
__device__ inline v16bf frag_f32_nt(const float* base, int ld, int lane) {
    int m  = lane & 15;
    int kb = (lane >> 4) << 3;
    const v4f* r  = (const v4f*)(base + m * ld + kb);
    const v4f* r2 = (const v4f*)(base + m * ld + kb + 16);
    v4f l0 = __builtin_nontemporal_load(r);
    v4f l1 = __builtin_nontemporal_load(r + 1);
    v4f h0 = __builtin_nontemporal_load(r2);
    v4f h1 = __builtin_nontemporal_load(r2 + 1);
    v16bf out;
#pragma unroll
    for (int i = 0; i < 4; ++i) {
        out[i]      = (bf16_t)l0[i];
        out[4 + i]  = (bf16_t)l1[i];
        out[8 + i]  = (bf16_t)h0[i];
        out[12 + i] = (bf16_t)h1[i];
    }
    return out;
}

// C/D (16x16 f32): vgpr v -> row v + (lane>=16)*8, col = lane&15
template <bool NT>
__device__ inline void store_tile(float* out, int ld, int lane, v8f acc, const float* bias_tile) {
    int col = lane & 15;
    int rb  = (lane >> 4) << 3;
    float b = bias_tile ? bias_tile[col] : 0.0f;
#pragma unroll
    for (int r = 0; r < 8; ++r) {
        float val = acc[r] + b;
        if (NT) __builtin_nontemporal_store(val, &out[(rb + r) * ld + col]);
        else    out[(rb + r) * ld + col] = val;
    }
}

// ---------------- GEMM kernels ----------------
// node projections: C(10000x1152) = s_bf16(10000x384) @ WnT' + bn
__global__ __launch_bounds__(32) void gemm_node_kernel(const bf16_t* __restrict__ A,
                                                       const bf16_t* __restrict__ WT,
                                                       const float* __restrict__ bias,
                                                       float* __restrict__ C) {
    int mt = blockIdx.x;            // 0..624
    int ng = blockIdx.y;            // 0..17 (4 n-tiles each)
    int lane = threadIdx.x;
    v8f acc[4] = {};
    const bf16_t* Abase = A + (size_t)mt * 16 * CS;
    for (int k0 = 0; k0 < CS; k0 += 32) {
        v16bf af = frag_bf16(Abase + k0, CS, lane);
#pragma unroll
        for (int j = 0; j < 4; ++j) {
            int n0 = (ng * 4 + j) * 16;
            v16bf bfr = frag_bf16(WT + (size_t)n0 * CS + k0, CS, lane);
            acc[j] = __builtin_amdgcn_wmma_f32_16x16x32_bf16(false, af, false, bfr,
                                                             (short)0, acc[j], false, false);
        }
    }
#pragma unroll
    for (int j = 0; j < 4; ++j) {
        int n0 = (ng * 4 + j) * 16;
        store_tile<false>(C + (size_t)mt * 16 * NPROJ + n0, NPROJ, lane, acc[j], bias + n0);
    }
}

// edge projections: zp(320000x48) = z(fp32 NT, cvt in regs) @ WzT' + bz
__global__ __launch_bounds__(32) void gemm_z_kernel(const float* __restrict__ Z,
                                                    const bf16_t* __restrict__ WT,
                                                    const float* __restrict__ bias,
                                                    float* __restrict__ C) {
    int mt = blockIdx.x;            // 0..19999
    int lane = threadIdx.x;
    v8f acc[3] = {};
    const float* Abase = Z + (size_t)mt * 16 * CZ;
    for (int k0 = 0; k0 < CZ; k0 += 32) {
        v16bf af = frag_f32_nt(Abase + k0, CZ, lane);
#pragma unroll
        for (int j = 0; j < 3; ++j) {
            v16bf bfr = frag_bf16(WT + (size_t)(j * 16) * CZ + k0, CZ, lane);
            acc[j] = __builtin_amdgcn_wmma_f32_16x16x32_bf16(false, af, false, bfr,
                                                             (short)0, acc[j], false, false);
        }
    }
#pragma unroll
    for (int j = 0; j < 3; ++j)
        store_tile<false>(C + (size_t)mt * 16 * ZCOLS + j * 16, ZCOLS, lane, acc[j], bias + j * 16);
}

// output: out(10000x384) = feats_bf16(10000x960) @ WoutT' + bout
// Final result is never re-read on device -> NT stores.
__global__ __launch_bounds__(32) void gemm_out_kernel(const bf16_t* __restrict__ A,
                                                      const bf16_t* __restrict__ WT,
                                                      const float* __restrict__ bias,
                                                      float* __restrict__ C) {
    int mt = blockIdx.x;            // 0..624
    int ng = blockIdx.y;            // 0..3 (6 n-tiles each -> 24 tiles = 384)
    int lane = threadIdx.x;
    v8f acc[6] = {};
    const bf16_t* Abase = A + (size_t)mt * 16 * FEATD;
    for (int k0 = 0; k0 < FEATD; k0 += 32) {
        v16bf af = frag_bf16(Abase + k0, FEATD, lane);
#pragma unroll
        for (int j = 0; j < 6; ++j) {
            int n0 = (ng * 6 + j) * 16;
            v16bf bfr = frag_bf16(WT + (size_t)n0 * FEATD + k0, FEATD, lane);
            acc[j] = __builtin_amdgcn_wmma_f32_16x16x32_bf16(false, af, false, bfr,
                                                             (short)0, acc[j], false, false);
        }
    }
#pragma unroll
    for (int j = 0; j < 6; ++j) {
        int n0 = (ng * 6 + j) * 16;
        store_tile<true>(C + (size_t)mt * 16 * 384 + n0, 384, lane, acc[j], bias + n0);
    }
}

// ---------------- prep / cast kernels ----------------
__global__ void prep_wn_kernel(const float* wq, const float* wkv, const float* wqp, const float* wkvp,
                               const float* bq, const float* bkv, const float* bqp, const float* bkvp,
                               bf16_t* WnT, float* bn) {
    int stride = gridDim.x * blockDim.x;
    for (int idx = blockIdx.x * blockDim.x + threadIdx.x; idx < NPROJ * CS; idx += stride) {
        int n = idx / CS, k = idx % CS;
        float w;
        if      (n < 192) w = wq  [k * 192 + n];
        else if (n < 576) w = wkv [k * 384 + (n - 192)];
        else if (n < 720) w = wqp [k * 144 + (n - 576)];
        else              w = wkvp[k * 432 + (n - 720)];
        WnT[(size_t)n * CS + k] = (bf16_t)w;
        if (k == 0) {
            float b = (n < 192) ? bq[n] : (n < 576) ? bkv[n - 192]
                      : (n < 720) ? bqp[n - 576] : bkvp[n - 720];
            bn[n] = b;
        }
    }
}

__global__ void prep_wz_kernel(const float* wb, const float* wdz,
                               const float* bb, const float* bdz,
                               bf16_t* WzT, float* bz) {
    int stride = gridDim.x * blockDim.x;
    for (int idx = blockIdx.x * blockDim.x + threadIdx.x; idx < ZCOLS * CZ; idx += stride) {
        int n = idx / CZ, k = idx % CZ;
        float w = 0.0f;
        if (n < 12)       w = wb [k * 12 + n];
        else if (n < 44)  w = wdz[k * 32 + (n - 12)];
        WzT[n * CZ + k] = (bf16_t)w;
        if (k == 0) bz[n] = (n < 12) ? bb[n] : (n < 44) ? bdz[n - 12] : 0.0f;
    }
}

__global__ void prep_wout_kernel(const float* wout, bf16_t* WoutT) {
    int stride = gridDim.x * blockDim.x;
    for (int idx = blockIdx.x * blockDim.x + threadIdx.x; idx < 384 * FEATD; idx += stride) {
        int n = idx / FEATD, k = idx % FEATD;
        WoutT[(size_t)n * FEATD + k] = (bf16_t)wout[(size_t)k * 384 + n];
    }
}

// s is read once here (s16 is what the GEMM consumes) -> NT load.
__global__ void cast_s_kernel(const float* s, bf16_t* s16) {
    int stride = gridDim.x * blockDim.x;
    for (int idx = blockIdx.x * blockDim.x + threadIdx.x; idx < N_RESC * CS; idx += stride)
        s16[idx] = (bf16_t)__builtin_nontemporal_load(&s[idx]);
}

// ---------------- ordered-float atomic-max encoding ----------------
__device__ inline unsigned fenc(float f) {
    unsigned u = __float_as_uint(f);
    return (u & 0x80000000u) ? ~u : (u | 0x80000000u);
}
__device__ inline float fdec(unsigned u) {
    u = (u & 0x80000000u) ? (u & 0x7fffffffu) : ~u;
    return __uint_as_float(u);
}

__global__ void init_accum_kernel(unsigned* mEnc, float* denom,
                                  float* o_acc, float* opt_acc, float* opair_acc) {
    int stride = gridDim.x * blockDim.x;
    int t = blockIdx.x * blockDim.x + threadIdx.x;
    for (int i = t; i < N_RESC * NH * 32;      i += stride) opair_acc[i] = 0.0f;
    for (int i = t; i < N_RESC * NH * PVn * 3; i += stride) opt_acc[i]  = 0.0f;
    for (int i = t; i < N_RESC * NH * CH;      i += stride) o_acc[i]    = 0.0f;
    for (int i = t; i < N_RESC * NH; i += stride) { denom[i] = 0.0f; mEnc[i] = fenc(-3.0e38f); }
}

// ---------------- geometry: rotate projected points into world frame -------
__global__ void points_kernel(const float* __restrict__ np, const float* __restrict__ rot,
                              const float* __restrict__ trans,
                              float* __restrict__ qpts, float* __restrict__ kpts,
                              float* __restrict__ vpts) {
    int idx = blockIdx.x * blockDim.x + threadIdx.x;
    if (idx >= N_RESC * NH * 16) return;
    int n = idx / (NH * 16);
    int r = idx % (NH * 16);
    int h = r / 16, p = r % 16;
    const float* R = rot + (size_t)n * 9;
    const float* T = trans + (size_t)n * 3;
    float x0, x1, x2;
    if (p < PQn) {  // q_pts: cols 576 + j*48 + h*4 + p
        const float* row = np + (size_t)n * NPROJ + 576 + h * 4 + p;
        x0 = row[0]; x1 = row[48]; x2 = row[96];
    } else {        // kv_pts: cols 720 + j*144 + h*12 + pp
        int pp = p - PQn;
        const float* row = np + (size_t)n * NPROJ + 720 + h * 12 + pp;
        x0 = row[0]; x1 = row[144]; x2 = row[288];
    }
    float y0 = R[0] * x0 + R[1] * x1 + R[2] * x2 + T[0];
    float y1 = R[3] * x0 + R[4] * x1 + R[5] * x2 + T[1];
    float y2 = R[6] * x0 + R[7] * x1 + R[8] * x2 + T[2];
    if (p < PQn) {
        float* o = qpts + ((size_t)(n * NH + h) * PQn + p) * 3;
        o[0] = y0; o[1] = y1; o[2] = y2;
    } else if (p - PQn < PQn) {
        float* o = kpts + ((size_t)(n * NH + h) * PQn + (p - PQn)) * 3;
        o[0] = y0; o[1] = y1; o[2] = y2;
    } else {
        float* o = vpts + ((size_t)(n * NH + h) * PVn + (p - 2 * PQn)) * 3;
        o[0] = y0; o[1] = y1; o[2] = y2;
    }
}

// ---------------- edge passes ----------------
__global__ void edge_logits_kernel(const float* __restrict__ np, const float* __restrict__ zp,
                                   const float* __restrict__ qpts, const float* __restrict__ kpts,
                                   const float* __restrict__ head_w, const float* __restrict__ mask,
                                   const int* __restrict__ ei, float* __restrict__ alog,
                                   unsigned* __restrict__ mEnc) {
    int idx = blockIdx.x * blockDim.x + threadIdx.x;
    if (idx >= N_EDGEC * NH) return;
    int e = idx / NH, h = idx % NH;
    int sN = ei[e], dN = ei[N_EDGEC + e];
    const float* q = np + (size_t)dN * NPROJ + h * CH;            // q block
    const float* k = np + (size_t)sN * NPROJ + 192 + h * 32;      // k block
    float dot = 0.0f;
#pragma unroll
    for (int c = 0; c < CH; ++c) dot += q[c] * k[c];
    float a = dot * 0.14433756729740643f               // sqrt(1/(3*C_H))
            + 0.5773502691896258f * zp[(size_t)e * ZCOLS + h];
    // point attention
    float hwv = head_w[h];
    float sp  = (hwv > 30.0f) ? hwv : log1pf(expf(hwv));
    float hw  = sp * 0.13608276348795434f;             // sqrt(1/(3*(PQ*9/2)))
    const float* qp = qpts + (size_t)(dN * NH + h) * PQn * 3;
    const float* kp = kpts + (size_t)(sN * NH + h) * PQn * 3;
    float ss = 0.0f;
#pragma unroll
    for (int i = 0; i < PQn * 3; ++i) { float d = qp[i] - kp[i]; ss += d * d; }
    a += -0.5f * ss * hw;
    a += INF_F * (mask[sN] * mask[dN] - 1.0f);
    alog[idx] = a;
    atomicMax(&mEnc[dN * NH + h], fenc(a));
}

__global__ void edge_exp_kernel(const int* __restrict__ ei, const unsigned* __restrict__ mEnc,
                                float* __restrict__ ea, float* __restrict__ denom) {
    int idx = blockIdx.x * blockDim.x + threadIdx.x;
    if (idx >= N_EDGEC * NH) return;
    int e = idx / NH;
    int h = idx % NH;
    int dN = ei[N_EDGEC + e];
    float m = fdec(mEnc[dN * NH + h]);
    float v = expf(ea[idx] - m);
    ea[idx] = v;
    atomicAdd(&denom[dN * NH + h], v);
}

__global__ void edge_accum_kernel(const int* __restrict__ ei, const float* __restrict__ ea,
                                  const float* __restrict__ np, const float* __restrict__ vpts,
                                  const float* __restrict__ zp,
                                  float* __restrict__ o_acc, float* __restrict__ opt_acc,
                                  float* __restrict__ opair_acc) {
    int idx = blockIdx.x * blockDim.x + threadIdx.x;
    if (idx >= N_EDGEC * NH) return;
    int e = idx / NH, h = idx % NH;
    int sN = ei[e], dN = ei[N_EDGEC + e];
    float w = ea[idx];
    const float* v = np + (size_t)sN * NPROJ + 192 + h * 32 + CH;     // v block
    float* oa = o_acc + (size_t)(dN * NH + h) * CH;
#pragma unroll
    for (int c = 0; c < CH; ++c) atomicAdd(&oa[c], w * v[c]);
    const float* vp = vpts + (size_t)(sN * NH + h) * PVn * 3;
    float* pa = opt_acc + (size_t)(dN * NH + h) * PVn * 3;
#pragma unroll
    for (int i = 0; i < PVn * 3; ++i) atomicAdd(&pa[i], w * vp[i]);
    const float* pz = zp + (size_t)e * ZCOLS + 12;
    float* qa = opair_acc + (size_t)(dN * NH + h) * 32;
#pragma unroll
    for (int c = 0; c < 32; ++c) atomicAdd(&qa[c], w * pz[c]);
}

// ---------------- node finalize: normalize, local frame, pack bf16 feats ---
__global__ void finalize_kernel(const float* __restrict__ denom, const float* __restrict__ o_acc,
                                const float* __restrict__ opt_acc, const float* __restrict__ opair_acc,
                                const float* __restrict__ rot, const float* __restrict__ trans,
                                bf16_t* __restrict__ feats) {
    int idx = blockIdx.x * blockDim.x + threadIdx.x;
    if (idx >= N_RESC * NH) return;
    int n = idx / NH, h = idx % NH;
    float inv = 1.0f / (denom[idx] + 1e-16f);
    bf16_t* f = feats + (size_t)n * FEATD;
    const float* oa = o_acc + (size_t)idx * CH;
#pragma unroll
    for (int c = 0; c < CH; ++c) f[h * CH + c] = (bf16_t)(oa[c] * inv);
    const float* R = rot + (size_t)n * 9;
    const float* T = trans + (size_t)n * 3;
    const float* pa = opt_acc + (size_t)idx * PVn * 3;
#pragma unroll
    for (int p = 0; p < PVn; ++p) {
        float g0 = pa[p * 3 + 0] * inv - T[0];
        float g1 = pa[p * 3 + 1] * inv - T[1];
        float g2 = pa[p * 3 + 2] * inv - T[2];
        // local = R^T * g
        float l0 = R[0] * g0 + R[3] * g1 + R[6] * g2;
        float l1 = R[1] * g0 + R[4] * g1 + R[7] * g2;
        float l2 = R[2] * g0 + R[5] * g1 + R[8] * g2;
        int pi = h * PVn + p;
        f[192 + pi]       = (bf16_t)l0;
        f[192 + 96 + pi]  = (bf16_t)l1;
        f[192 + 192 + pi] = (bf16_t)l2;
        f[480 + pi]       = (bf16_t)sqrtf(l0 * l0 + l1 * l1 + l2 * l2 + 1e-8f);
    }
    const float* qa = opair_acc + (size_t)idx * 32;
#pragma unroll
    for (int c = 0; c < 32; ++c) f[576 + h * 32 + c] = (bf16_t)(qa[c] * inv);
}

// ---------------- launcher ----------------
extern "C" void kernel_launch(void* const* d_in, const int* in_sizes, int n_in,
                              void* d_out, int out_size, void* d_ws, size_t ws_size,
                              hipStream_t stream) {
    (void)in_sizes; (void)n_in; (void)out_size; (void)ws_size;
    const float* s     = (const float*)d_in[0];
    const float* z     = (const float*)d_in[1];
    const float* rot   = (const float*)d_in[2];
    const float* trans = (const float*)d_in[3];
    const float* mask  = (const float*)d_in[4];
    const int*   ei    = (const int*)  d_in[5];
    const float* wq    = (const float*)d_in[6];
    const float* bq    = (const float*)d_in[7];
    const float* wkv   = (const float*)d_in[8];
    const float* bkv   = (const float*)d_in[9];
    const float* wqp   = (const float*)d_in[10];
    const float* bqp   = (const float*)d_in[11];
    const float* wkvp  = (const float*)d_in[12];
    const float* bkvp  = (const float*)d_in[13];
    const float* wb    = (const float*)d_in[14];
    const float* bb    = (const float*)d_in[15];
    const float* wdz   = (const float*)d_in[16];
    const float* bdz   = (const float*)d_in[17];
    const float* hwp   = (const float*)d_in[18];
    const float* wout  = (const float*)d_in[19];
    const float* bout  = (const float*)d_in[20];
    float* out = (float*)d_out;

    char* base = (char*)d_ws;
    size_t off = 0;
    auto alloc = [&](size_t bytes) -> void* {
        void* p = base + off;
        off += (bytes + 255) & ~(size_t)255;
        return p;
    };
    bf16_t*  s16      = (bf16_t*)alloc((size_t)N_RESC * CS * 2);
    bf16_t*  WnT      = (bf16_t*)alloc((size_t)NPROJ * CS * 2);
    float*   bn       = (float*) alloc(NPROJ * 4);
    bf16_t*  WzT      = (bf16_t*)alloc((size_t)ZCOLS * CZ * 2);
    float*   bz       = (float*) alloc(ZCOLS * 4);
    bf16_t*  WoutT    = (bf16_t*)alloc((size_t)384 * FEATD * 2);
    float*   nodep    = (float*) alloc((size_t)N_RESC * NPROJ * 4);
    float*   qpts     = (float*) alloc((size_t)N_RESC * NH * PQn * 3 * 4);
    float*   kpts     = (float*) alloc((size_t)N_RESC * NH * PQn * 3 * 4);
    float*   vpts     = (float*) alloc((size_t)N_RESC * NH * PVn * 3 * 4);
    float*   zp       = (float*) alloc((size_t)N_EDGEC * ZCOLS * 4);
    float*   ea       = (float*) alloc((size_t)N_EDGEC * NH * 4);
    unsigned* mEnc    = (unsigned*)alloc((size_t)N_RESC * NH * 4);
    float*   denom    = (float*) alloc((size_t)N_RESC * NH * 4);
    float*   o_acc    = (float*) alloc((size_t)N_RESC * NH * CH * 4);
    float*   opt_acc  = (float*) alloc((size_t)N_RESC * NH * PVn * 3 * 4);
    float*   opair    = (float*) alloc((size_t)N_RESC * NH * 32 * 4);
    bf16_t*  feats    = (bf16_t*)alloc((size_t)N_RESC * FEATD * 2);

    const int B = 256;
    // weight prep + casts
    prep_wn_kernel  <<<(NPROJ * CS + B - 1) / B, B, 0, stream>>>(wq, wkv, wqp, wkvp,
                                                                 bq, bkv, bqp, bkvp, WnT, bn);
    prep_wz_kernel  <<<(ZCOLS * CZ + B - 1) / B, B, 0, stream>>>(wb, wdz, bb, bdz, WzT, bz);
    prep_wout_kernel<<<(384 * FEATD + B - 1) / B, B, 0, stream>>>(wout, WoutT);
    cast_s_kernel   <<<(N_RESC * CS + B - 1) / B, B, 0, stream>>>(s, s16);
    init_accum_kernel<<<15000, B, 0, stream>>>(mEnc, denom, o_acc, opt_acc, opair);

    // GEMMs (WMMA bf16, fp32 accumulate)
    gemm_node_kernel<<<dim3(N_RESC / 16, NPROJ / 64), 32, 0, stream>>>(s16, WnT, bn, nodep);
    gemm_z_kernel   <<<N_EDGEC / 16, 32, 0, stream>>>(z, WzT, bz, zp);

    // geometry
    points_kernel<<<(N_RESC * NH * 16 + B - 1) / B, B, 0, stream>>>(nodep, rot, trans,
                                                                    qpts, kpts, vpts);
    // edge passes
    int ETH = N_EDGEC * NH;
    edge_logits_kernel<<<(ETH + B - 1) / B, B, 0, stream>>>(nodep, zp, qpts, kpts,
                                                            hwp, mask, ei, ea, mEnc);
    edge_exp_kernel   <<<(ETH + B - 1) / B, B, 0, stream>>>(ei, mEnc, ea, denom);
    edge_accum_kernel <<<(ETH + B - 1) / B, B, 0, stream>>>(ei, ea, nodep, vpts, zp,
                                                            o_acc, opt_acc, opair);
    // node finalize + output projection
    finalize_kernel<<<(N_RESC * NH + B - 1) / B, B, 0, stream>>>(denom, o_acc, opt_acc, opair,
                                                                 rot, trans, feats);
    gemm_out_kernel<<<dim3(N_RESC / 16, 4), 32, 0, stream>>>(feats, WoutT, bout, out);
}
